// TransformerSentenceEncoderLayer_1683627180629
// MI455X (gfx1250) — 8-bit matrix operand order repaired and checked by reading, compiled, not run on hardware
//
#include <hip/hip_runtime.h>
#include <math.h>

#define T_   512
#define B_   8
#define D_   768
#define H_   12
#define HD_  64
#define FF_  3072
#define NTOK (T_ * B_)
#define BH   (B_ * H_)

typedef int v8i __attribute__((ext_vector_type(8)));

enum {
  SLOT_X = 0, SLOT_Q, SLOT_K, SLOT_V, SLOT_EXP, SLOT_CTX,
  SLOT_RES1, SLOT_LN1, SLOT_RELU, SLOT_RES2, NSLOT
};


__device__ __forceinline__ v8i vzero() {
  v8i z = {0, 0, 0, 0, 0, 0, 0, 0};
  return z;
}

__device__ __forceinline__ float slot_scale(const unsigned* slots, int idx, float n) {
  return fmaxf(__uint_as_float(slots[idx]), 1e-8f) / n;
}

__device__ __forceinline__ float rq_ratio(float a, float b) {
  int e;
  float m = frexpf(a / b, &e);
  return roundf(m * 2147483648.0f) * exp2f((float)(e - 31));
}

__device__ __forceinline__ void amax_commit(float v, unsigned* slot) {
  __shared__ float red[256];
  red[threadIdx.x] = v;
  __syncthreads();
  for (int s = blockDim.x >> 1; s > 0; s >>= 1) {
    if ((int)threadIdx.x < s) red[threadIdx.x] = fmaxf(red[threadIdx.x], red[threadIdx.x + s]);
    __syncthreads();
  }
  if (threadIdx.x == 0) atomicMax(slot, __float_as_uint(red[0]));
}

__device__ __forceinline__ v8i ldA(const signed char* p, int stride, int lane) {
  const int m = lane & 15, half = lane >> 4;
  const signed char* q = p + (size_t)m * stride + (half << 4);
  v8i f;
#pragma unroll
  for (int j = 0; j < 8; ++j) {
    const int kb = ((j >> 2) << 5) + ((j & 3) << 2);
    f[j] = *(const int*)(q + kb);
  }
  return f;
}

__device__ __forceinline__ v8i ldB(const signed char* p, int stride, int lane) {
  const int n = lane & 15, half = lane >> 4;
  const signed char* q = p + (size_t)n * stride + (half << 4);
  v8i f;
#pragma unroll
  for (int j = 0; j < 8; ++j) {
    const int kb = ((j >> 2) << 5) + ((j & 3) << 2);
    f[j] = *(const int*)(q + kb);
  }
  return f;
}


__global__ __launch_bounds__(64) void k_zero_slots(unsigned* slots) {
  if (threadIdx.x < NSLOT) slots[threadIdx.x] = 0u;
}

__global__ __launch_bounds__(256)
void k_quant_weight(const float* __restrict__ W, int I,
                    signed char* __restrict__ w8, float* __restrict__ wscale) {
  __shared__ float sh[256];
  const int o = blockIdx.x;
  const float* wr = W + (size_t)o * I;
  float lm = 0.f;
  for (int i = threadIdx.x; i < I; i += 256) lm = fmaxf(lm, fabsf(wr[i]));
  sh[threadIdx.x] = lm;
  __syncthreads();
  for (int s = 128; s > 0; s >>= 1) {
    if ((int)threadIdx.x < s) sh[threadIdx.x] = fmaxf(sh[threadIdx.x], sh[threadIdx.x + s]);
    __syncthreads();
  }
  const float sc = fmaxf(sh[0], 1e-8f) / 127.f;
  if (threadIdx.x == 0) wscale[o] = sc;
  for (int i = threadIdx.x; i < I; i += 256) {
    float q = fminf(fmaxf(roundf(wr[i] / sc), -128.f), 127.f);
    w8[(size_t)o * I + i] = (signed char)(int)q;
  }
}

__global__ __launch_bounds__(256)
void k_absmax_f32(const float* __restrict__ x, size_t n, unsigned* __restrict__ slot) {
  float lm = 0.f;
  for (size_t i = blockIdx.x * (size_t)blockDim.x + threadIdx.x; i < n;
       i += (size_t)gridDim.x * blockDim.x)
    lm = fmaxf(lm, fabsf(x[i]));
  amax_commit(lm, slot);
}

__global__ __launch_bounds__(256)
void k_quant_input(const float* __restrict__ x, const float* __restrict__ xsf,
                   const unsigned* __restrict__ slots,
                   signed char* __restrict__ out, size_t n) {
  const float pre = xsf[0];
  const float s = slot_scale(slots, SLOT_X, 127.f);
  const float ratio = rq_ratio(pre, s);
  for (size_t i = blockIdx.x * (size_t)blockDim.x + threadIdx.x; i < n;
       i += (size_t)gridDim.x * blockDim.x) {
    float z = roundf(x[i] / pre);
    float o = fminf(fmaxf(roundf(z * ratio), -128.f), 127.f);
    out[i] = (signed char)(int)o;
  }
}

__global__ __launch_bounds__(128)
void k_gemm_s8(const signed char* __restrict__ A, int lda,
               const signed char* __restrict__ Bm, int ldb,
               int* __restrict__ C, int ldc, int K,
               const float* __restrict__ bias,
               const float* __restrict__ wscale,
               const unsigned* __restrict__ slots, int xslot, float xn) {
  const int wave = threadIdx.x >> 5, lane = threadIdx.x & 31;
  const int row0 = blockIdx.x * 64 + wave * 16;
  const int n0 = blockIdx.y * 128;
  const int nn = lane & 15, half = lane >> 4;
  v8i acc[8] = {vzero(), vzero(), vzero(), vzero(),
                vzero(), vzero(), vzero(), vzero()};
  for (int k0 = 0; k0 < K; k0 += 64) {
    v8i a = ldA(A + (size_t)row0 * lda + k0, lda, lane);
    if (k0 + 64 < K)
      __builtin_prefetch(A + (size_t)row0 * lda + k0 + 64, 0, 0);
#pragma unroll
    for (int j = 0; j < 8; ++j) {
      v8i b = ldB(Bm + (size_t)(n0 + j * 16) * ldb + k0, ldb, lane);
      acc[j] = __builtin_amdgcn_wmma_i32_16x16x64_iu8(true, a, true, b, acc[j], false, false);
    }
  }
  const float xs = slot_scale(slots, xslot, xn);
#pragma unroll
  for (int j = 0; j < 8; ++j) {
    const int col = n0 + j * 16 + nn;
    const float bs = wscale[col] * xs;
    float bf = fminf(fmaxf(roundf(bias[col] / bs), -2147483648.0f), 2147483520.0f);
    const int bi = (int)(long long)bf;
#pragma unroll
    for (int r = 0; r < 8; ++r)
      C[(size_t)(row0 + r + half * 8) * ldc + col] = acc[j][r] + bi;
  }
}

__global__ __launch_bounds__(256)
void k_absmax_acc(const int* __restrict__ acc, size_t M, int Nc, int ldc,
                  const float* __restrict__ wscale,
                  const unsigned* __restrict__ slots, int xslot, float xn,
                  float extra, int relu, unsigned* __restrict__ outslot) {
  const float xs = slot_scale(slots, xslot, xn);
  const size_t n = M * (size_t)Nc;
  float lm = 0.f;
  for (size_t i = blockIdx.x * (size_t)blockDim.x + threadIdx.x; i < n;
       i += (size_t)gridDim.x * blockDim.x) {
    const int col = (int)(i % Nc);
    const size_t row = i / Nc;
    const float pre = (wscale ? wscale[col] : 1.f) * xs * extra;
    float v = (float)acc[row * ldc + col] * pre;
    if (relu) v = fmaxf(v, 0.f);
    lm = fmaxf(lm, fabsf(v));
  }
  amax_commit(lm, outslot);
}

__global__ __launch_bounds__(256)
void k_requant8(const int* __restrict__ acc, size_t M, int Nc, int ldc,
                const float* __restrict__ wscale,
                const unsigned* __restrict__ slots, int xslot, float xn,
                float extra, int amaxslot, int relu_direct,
                signed char* __restrict__ out) {
  const float xs = slot_scale(slots, xslot, xn);
  const float s = slot_scale(slots, amaxslot, 127.f);
  const size_t n = M * (size_t)Nc;
  for (size_t i = blockIdx.x * (size_t)blockDim.x + threadIdx.x; i < n;
       i += (size_t)gridDim.x * blockDim.x) {
    const int col = (int)(i % Nc);
    const size_t row = i / Nc;
    const float pre = (wscale ? wscale[col] : 1.f) * xs * extra;
    float o;
    if (relu_direct)
      o = roundf(fmaxf((float)acc[row * ldc + col] * pre, 0.f) / s);
    else
      o = roundf((float)acc[row * ldc + col] * rq_ratio(pre, s));
    out[i] = (signed char)(int)fminf(fmaxf(o, -128.f), 127.f);
  }
}

__global__ __launch_bounds__(256)
void k_absmax_res(const int* __restrict__ acc, const float* __restrict__ wscale,
                  const unsigned* __restrict__ slots, int xslot,
                  const signed char* __restrict__ res8, int resslot,
                  size_t M, int Nc, unsigned* __restrict__ outslot) {
  const float xs = slot_scale(slots, xslot, 127.f);
  const float rs = slot_scale(slots, resslot, 127.f);
  const size_t n = M * (size_t)Nc;
  float lm = 0.f;
  for (size_t i = blockIdx.x * (size_t)blockDim.x + threadIdx.x; i < n;
       i += (size_t)gridDim.x * blockDim.x) {
    const int col = (int)(i % Nc);
    float v = (float)acc[i] * wscale[col] * xs + (float)res8[i] * rs;
    lm = fmaxf(lm, fabsf(v));
  }
  amax_commit(lm, outslot);
}

__global__ __launch_bounds__(256)
void k_requant_res16(const int* __restrict__ acc, const float* __restrict__ wscale,
                     const unsigned* __restrict__ slots, int xslot,
                     const signed char* __restrict__ res8, int resslot,
                     int amaxslot, size_t M, int Nc, short* __restrict__ out) {
  const float xs = slot_scale(slots, xslot, 127.f);
  const float rs = slot_scale(slots, resslot, 127.f);
  const float s = slot_scale(slots, amaxslot, 32767.f);
  const float rratio = rq_ratio(rs, s);
  const size_t n = M * (size_t)Nc;
  for (size_t i = blockIdx.x * (size_t)blockDim.x + threadIdx.x; i < n;
       i += (size_t)gridDim.x * blockDim.x) {
    const int col = (int)(i % Nc);
    const float pre = wscale[col] * xs;
    float o = roundf((float)acc[i] * rq_ratio(pre, s)) + roundf((float)res8[i] * rratio);
    out[i] = (short)(int)fminf(fmaxf(o, -32768.f), 32767.f);
  }
}

__global__ __launch_bounds__(256)
void k_layernorm(const short* __restrict__ xin,
                 const float* __restrict__ weight, const float* __restrict__ bias,
                 float* __restrict__ yout, unsigned* __restrict__ amaxslot,
                 float* __restrict__ fout, float* __restrict__ fscale) {
  __shared__ float sh[256];
  __shared__ float bc[2];
  const int row = blockIdx.x;
  const float base = sqrtf((float)D_) / 1073741824.0f;
  const short* xr = xin + (size_t)row * D_;
  float ls = 0.f;
  for (int c = threadIdx.x; c < D_; c += 256) ls += (float)xr[c];
  sh[threadIdx.x] = ls;
  __syncthreads();
  for (int s = 128; s > 0; s >>= 1) {
    if ((int)threadIdx.x < s) sh[threadIdx.x] += sh[threadIdx.x + s];
    __syncthreads();
  }
  if (threadIdx.x == 0) bc[0] = roundf(sh[0] / (float)D_);
  __syncthreads();
  const float mean = bc[0];
  float lv = 0.f;
  for (int c = threadIdx.x; c < D_; c += 256) {
    float y = (float)xr[c] - mean;
    lv += y * y;
  }
  __syncthreads();
  sh[threadIdx.x] = lv;
  __syncthreads();
  for (int s = 128; s > 0; s >>= 1) {
    if ((int)threadIdx.x < s) sh[threadIdx.x] += sh[threadIdx.x + s];
    __syncthreads();
  }
  if (threadIdx.x == 0) {
    float stdv = fmaxf(floorf(sqrtf(sh[0])), 1.f);
    bc[1] = floorf(2147483648.0f / stdv);
  }
  __syncthreads();
  const float factor = bc[1];
  float lmax = 0.f;
  for (int c = threadIdx.x; c < D_; c += 256) {
    float y = (float)xr[c] - mean;
    float y2 = floorf(y * factor * 0.5f);
    y2 += floorf((bias[c] / weight[c]) / base);
    if (fout) {
      fout[(size_t)row * D_ + c] = y2 * base * weight[c];
    } else {
      yout[(size_t)row * D_ + c] = y2;
      lmax = fmaxf(lmax, fabsf(y2 * base * weight[c]));
    }
  }
  if (amaxslot) amax_commit(lmax, amaxslot);
  if (fscale && row == 0)
    for (int c = threadIdx.x; c < D_; c += 256) fscale[c] = base * weight[c];
}

__global__ __launch_bounds__(256)
void k_requant_ln8(const float* __restrict__ y, const float* __restrict__ weight,
                   const unsigned* __restrict__ slots, int amaxslot,
                   signed char* __restrict__ out, size_t M) {
  const float base = sqrtf((float)D_) / 1073741824.0f;
  const float s = slot_scale(slots, amaxslot, 127.f);
  const size_t n = M * (size_t)D_;
  for (size_t i = blockIdx.x * (size_t)blockDim.x + threadIdx.x; i < n;
       i += (size_t)gridDim.x * blockDim.x) {
    const int c = (int)(i % D_);
    float o = roundf(y[i] * rq_ratio(base * weight[c], s));
    out[i] = (signed char)(int)fminf(fmaxf(o, -128.f), 127.f);
  }
}

__global__ __launch_bounds__(256)
void k_attn_scores_exp(const signed char* __restrict__ q8,
                       const signed char* __restrict__ k8,
                       float* __restrict__ expbuf, unsigned* __restrict__ slots) {
  __shared__ int sc[16][512];
  __shared__ float rmx[16][16];
  __shared__ float rowmax[16];
  const int bh = blockIdx.x, b = bh / H_, h = bh % H_;
  const int t0 = blockIdx.y * 16;
  const int wave = threadIdx.x >> 5, lane = threadIdx.x & 31;
  const int s0 = wave * 64;
  const int nn = lane & 15, half = lane >> 4;
  v8i a = ldA(q8 + ((size_t)t0 * B_ + b) * D_ + h * HD_, B_ * D_, lane);
#pragma unroll
  for (int j = 0; j < 4; ++j) {
    v8i bf = ldB(k8 + ((size_t)(s0 + j * 16) * B_ + b) * D_ + h * HD_, B_ * D_, lane);
    v8i c = vzero();
    c = __builtin_amdgcn_wmma_i32_16x16x64_iu8(true, a, true, bf, c, false, false);
#pragma unroll
    for (int r = 0; r < 8; ++r) sc[r + half * 8][s0 + j * 16 + nn] = c[r];
  }
  __syncthreads();
  const float scale = slot_scale(slots, SLOT_Q, 127.f) * slot_scale(slots, SLOT_K, 127.f);
  {
    const int r = threadIdx.x >> 4, tc = threadIdx.x & 15;
    int mx = -2147483647;
    for (int s = tc; s < 512; s += 16) mx = max(mx, sc[r][s]);
    rmx[r][tc] = (float)mx;
  }
  __syncthreads();
  if (threadIdx.x < 16) {
    float m2 = rmx[threadIdx.x][0];
    for (int i = 1; i < 16; ++i) m2 = fmaxf(m2, rmx[threadIdx.x][i]);
    rowmax[threadIdx.x] = m2;
  }
  __syncthreads();
  const float x0i = floorf(-0.6931f / scale);
  const float bint = floorf(0.96963238f / 0.35815147f / scale);
  const float cint = floorf(1.0f / (0.35815147f * scale * scale));
  const float escale = 0.35815147f * scale * scale * (1.0f / 1073741824.0f);
  float lmax = 0.f;
  for (int idx = threadIdx.x; idx < 16 * 512; idx += 256) {
    const int rr = idx >> 9, ss = idx & 511;
    float xi = (float)sc[rr][ss] - rowmax[rr];
    xi = fmaxf(xi, 30.f * x0i);
    float qq = floorf(xi / x0i);
    float rres = xi - x0i * qq;
    float ei = rres * (rres + bint) + cint;
    ei = fmaxf(floorf(ei * exp2f(30.f - qq)), 0.f);
    float ev = ei * escale;
    expbuf[((size_t)bh * T_ + t0 + rr) * T_ + ss] = ev;
    lmax = fmaxf(lmax, ev);
  }
  amax_commit(lmax, &slots[SLOT_EXP]);
}

__global__ __launch_bounds__(256)
void k_softmax_norm(const float* __restrict__ expbuf,
                    unsigned char* __restrict__ probs,
                    const unsigned* __restrict__ slots) {
  __shared__ float e2[512];
  __shared__ float sh[256];
  const size_t row = blockIdx.x;
  const float scale = slot_scale(slots, SLOT_Q, 127.f) * slot_scale(slots, SLOT_K, 127.f);
  const float escale = 0.35815147f * scale * scale * (1.0f / 1073741824.0f);
  const float s16 = slot_scale(slots, SLOT_EXP, 32767.f);
  const float ratio = rq_ratio(escale, s16);
  float lsum = 0.f;
  for (int s = threadIdx.x; s < 512; s += 256) {
    float z = roundf(expbuf[row * 512 + s] / escale);
    float v = fminf(fmaxf(roundf(z * ratio), -32768.f), 32767.f);
    e2[s] = v;
    lsum += v;
  }
  sh[threadIdx.x] = lsum;
  __syncthreads();
  for (int t = 128; t > 0; t >>= 1) {
    if ((int)threadIdx.x < t) sh[threadIdx.x] += sh[threadIdx.x + t];
    __syncthreads();
  }
  const float total = fmaxf(sh[0], 1.f);
  const float factor = floorf(4294967296.0f / total);
  for (int s = threadIdx.x; s < 512; s += 256) {
    float oi = floorf(e2[s] * factor * (1.0f / 16777216.0f));
    probs[row * 512 + s] = (unsigned char)(int)fminf(fmaxf(oi, 0.f), 255.f);
  }
}

__global__ __launch_bounds__(32)
void k_ctx_gemm(const unsigned char* __restrict__ probs,
                const signed char* __restrict__ v8,
                int* __restrict__ ctxacc) {
  const int bh = blockIdx.x, b = bh / H_, h = bh % H_;
  const int t0 = blockIdx.y * 16;
  const int lane = threadIdx.x & 31;
  const int nn = lane & 15, half = lane >> 4;
  v8i acc[4] = {vzero(), vzero(), vzero(), vzero()};
  const signed char* Ab = (const signed char*)(probs + ((size_t)bh * T_ + t0) * T_);
  for (int k0 = 0; k0 < T_; k0 += 64) {
    v8i a = ldA(Ab + k0, T_, lane);
#pragma unroll
    for (int j = 0; j < 4; ++j) {
      const signed char* Vb = v8 + (size_t)b * D_ + h * HD_ + j * 16 + nn;
      v8i bf;
#pragma unroll
      for (int jj = 0; jj < 8; ++jj) {
        const int kb = ((jj >> 2) << 5) + (half << 4) + ((jj & 3) << 2);
        unsigned vv = 0;
#pragma unroll
        for (int i = 0; i < 4; ++i)
          vv |= ((unsigned)(unsigned char)Vb[(size_t)(k0 + kb + i) * (B_ * D_)]) << (8 * i);
        bf[jj] = (int)vv;
      }
      acc[j] = __builtin_amdgcn_wmma_i32_16x16x64_iu8(false, a, true, bf, acc[j], false, false);
    }
  }
#pragma unroll
  for (int j = 0; j < 4; ++j)
#pragma unroll
    for (int r = 0; r < 8; ++r)
      ctxacc[((size_t)(t0 + r + half * 8) * B_ + b) * D_ + h * HD_ + j * 16 + nn] = acc[j][r];
}


extern "C" void kernel_launch(void* const* d_in, const int* in_sizes, int n_in,
                              void* d_out, int out_size, void* d_ws, size_t ws_size,
                              hipStream_t stream) {
  (void)in_sizes; (void)n_in; (void)out_size; (void)ws_size;
  const float* x    = (const float*)d_in[0];
  const float* xsf  = (const float*)d_in[1];
  const float* Wq   = (const float*)d_in[2];  const float* bq  = (const float*)d_in[3];
  const float* Wk   = (const float*)d_in[4];  const float* bk  = (const float*)d_in[5];
  const float* Wv   = (const float*)d_in[6];  const float* bv  = (const float*)d_in[7];
  const float* Wo   = (const float*)d_in[8];  const float* bo  = (const float*)d_in[9];
  const float* ln1w = (const float*)d_in[10]; const float* ln1b = (const float*)d_in[11];
  const float* W1   = (const float*)d_in[12]; const float* b1  = (const float*)d_in[13];
  const float* W2   = (const float*)d_in[14]; const float* b2  = (const float*)d_in[15];
  const float* ln2w = (const float*)d_in[16]; const float* ln2b = (const float*)d_in[17];

  char* ws = (char*)d_ws;
  size_t off = 0;
  auto alloc = [&](size_t bytes) -> char* {
    off = (off + 255) & ~(size_t)255;
    char* p = ws + off;
    off += bytes;
    return p;
  };
  unsigned* slots = (unsigned*)alloc(64 * sizeof(unsigned));
  float* wsQ  = (float*)alloc(D_ * 4);
  float* wsK  = (float*)alloc(D_ * 4);
  float* wsV  = (float*)alloc(D_ * 4);
  float* wsO  = (float*)alloc(D_ * 4);
  float* wsF1 = (float*)alloc(FF_ * 4);
  float* wsF2 = (float*)alloc(D_ * 4);
  signed char* W8Q  = (signed char*)alloc((size_t)D_ * D_);
  signed char* W8K  = (signed char*)alloc((size_t)D_ * D_);
  signed char* W8V  = (signed char*)alloc((size_t)D_ * D_);
  signed char* W8O  = (signed char*)alloc((size_t)D_ * D_);
  signed char* W8F1 = (signed char*)alloc((size_t)FF_ * D_);
  signed char* W8F2 = (signed char*)alloc((size_t)D_ * FF_);
  signed char* X8   = (signed char*)alloc((size_t)NTOK * D_);
  signed char* Q8   = (signed char*)alloc((size_t)NTOK * D_);
  signed char* K8   = (signed char*)alloc((size_t)NTOK * D_);
  signed char* V8   = (signed char*)alloc((size_t)NTOK * D_);
  signed char* CTX8 = (signed char*)alloc((size_t)NTOK * D_);
  signed char* LN18 = (signed char*)alloc((size_t)NTOK * D_);
  signed char* RELU8= (signed char*)alloc((size_t)NTOK * FF_);
  int*   ACC   = (int*)alloc((size_t)NTOK * FF_ * 4);
  short* RES16 = (short*)alloc((size_t)NTOK * D_ * 2);
  float* YLN   = (float*)alloc((size_t)NTOK * D_ * 4);
  float* EXPB  = (float*)alloc((size_t)BH * T_ * T_ * 4);
  unsigned char* PRB = (unsigned char*)alloc((size_t)BH * T_ * T_);

  float* out = (float*)d_out;
  float* out_scale = out + (size_t)NTOK * D_;

  const dim3 blk256(256), blk128(128), blk32(32);
  const dim3 gEW(2048);
  const dim3 gGemmD(NTOK / 64, D_ / 128);
  const dim3 gGemmF(NTOK / 64, FF_ / 128);
  const dim3 gAttn(BH, T_ / 16);

  k_zero_slots<<<1, 64, 0, stream>>>(slots);

  k_quant_weight<<<D_,  blk256, 0, stream>>>(Wq, D_,  W8Q,  wsQ);
  k_quant_weight<<<D_,  blk256, 0, stream>>>(Wk, D_,  W8K,  wsK);
  k_quant_weight<<<D_,  blk256, 0, stream>>>(Wv, D_,  W8V,  wsV);
  k_quant_weight<<<D_,  blk256, 0, stream>>>(Wo, D_,  W8O,  wsO);
  k_quant_weight<<<FF_, blk256, 0, stream>>>(W1, D_,  W8F1, wsF1);
  k_quant_weight<<<D_,  blk256, 0, stream>>>(W2, FF_, W8F2, wsF2);

  k_absmax_f32<<<gEW, blk256, 0, stream>>>(x, (size_t)NTOK * D_, &slots[SLOT_X]);
  k_quant_input<<<gEW, blk256, 0, stream>>>(x, xsf, slots, X8, (size_t)NTOK * D_);

  k_gemm_s8<<<gGemmD, blk128, 0, stream>>>(X8, D_, W8Q, D_, ACC, D_, D_, bq, wsQ, slots, SLOT_X, 127.f);
  k_absmax_acc<<<gEW, blk256, 0, stream>>>(ACC, NTOK, D_, D_, wsQ, slots, SLOT_X, 127.f, 0.125f, 0, &slots[SLOT_Q]);
  k_requant8<<<gEW, blk256, 0, stream>>>(ACC, NTOK, D_, D_, wsQ, slots, SLOT_X, 127.f, 0.125f, SLOT_Q, 0, Q8);

  k_gemm_s8<<<gGemmD, blk128, 0, stream>>>(X8, D_, W8K, D_, ACC, D_, D_, bk, wsK, slots, SLOT_X, 127.f);
  k_absmax_acc<<<gEW, blk256, 0, stream>>>(ACC, NTOK, D_, D_, wsK, slots, SLOT_X, 127.f, 1.f, 0, &slots[SLOT_K]);
  k_requant8<<<gEW, blk256, 0, stream>>>(ACC, NTOK, D_, D_, wsK, slots, SLOT_X, 127.f, 1.f, SLOT_K, 0, K8);

  k_gemm_s8<<<gGemmD, blk128, 0, stream>>>(X8, D_, W8V, D_, ACC, D_, D_, bv, wsV, slots, SLOT_X, 127.f);
  k_absmax_acc<<<gEW, blk256, 0, stream>>>(ACC, NTOK, D_, D_, wsV, slots, SLOT_X, 127.f, 1.f, 0, &slots[SLOT_V]);
  k_requant8<<<gEW, blk256, 0, stream>>>(ACC, NTOK, D_, D_, wsV, slots, SLOT_X, 127.f, 1.f, SLOT_V, 0, V8);

  k_attn_scores_exp<<<gAttn, blk256, 0, stream>>>(Q8, K8, EXPB, slots);
  k_softmax_norm<<<BH * T_, blk256, 0, stream>>>(EXPB, PRB, slots);

  k_ctx_gemm<<<gAttn, blk32, 0, stream>>>(PRB, V8, ACC);
  k_absmax_acc<<<gEW, blk256, 0, stream>>>(ACC, NTOK, D_, D_, nullptr, slots, SLOT_V, 127.f, 1.f / 256.f, 0, &slots[SLOT_CTX]);
  k_requant8<<<gEW, blk256, 0, stream>>>(ACC, NTOK, D_, D_, nullptr, slots, SLOT_V, 127.f, 1.f / 256.f, SLOT_CTX, 0, CTX8);

  k_gemm_s8<<<gGemmD, blk128, 0, stream>>>(CTX8, D_, W8O, D_, ACC, D_, D_, bo, wsO, slots, SLOT_CTX, 127.f);
  k_absmax_res<<<gEW, blk256, 0, stream>>>(ACC, wsO, slots, SLOT_CTX, X8, SLOT_X, NTOK, D_, &slots[SLOT_RES1]);
  k_requant_res16<<<gEW, blk256, 0, stream>>>(ACC, wsO, slots, SLOT_CTX, X8, SLOT_X, SLOT_RES1, NTOK, D_, RES16);
  k_layernorm<<<NTOK, blk256, 0, stream>>>(RES16, ln1w, ln1b, YLN, &slots[SLOT_LN1], nullptr, nullptr);
  k_requant_ln8<<<gEW, blk256, 0, stream>>>(YLN, ln1w, slots, SLOT_LN1, LN18, NTOK);

  k_gemm_s8<<<gGemmF, blk128, 0, stream>>>(LN18, D_, W8F1, D_, ACC, FF_, D_, b1, wsF1, slots, SLOT_LN1, 127.f);
  k_absmax_acc<<<gEW, blk256, 0, stream>>>(ACC, NTOK, FF_, FF_, wsF1, slots, SLOT_LN1, 127.f, 1.f, 1, &slots[SLOT_RELU]);
  k_requant8<<<gEW, blk256, 0, stream>>>(ACC, NTOK, FF_, FF_, wsF1, slots, SLOT_LN1, 127.f, 1.f, SLOT_RELU, 1, RELU8);

  k_gemm_s8<<<gGemmD, blk128, 0, stream>>>(RELU8, FF_, W8F2, FF_, ACC, D_, FF_, b2, wsF2, slots, SLOT_RELU, 127.f);
  k_absmax_res<<<gEW, blk256, 0, stream>>>(ACC, wsF2, slots, SLOT_RELU, LN18, SLOT_LN1, NTOK, D_, &slots[SLOT_RES2]);
  k_requant_res16<<<gEW, blk256, 0, stream>>>(ACC, wsF2, slots, SLOT_RELU, LN18, SLOT_LN1, SLOT_RES2, NTOK, D_, RES16);

  k_layernorm<<<NTOK, blk256, 0, stream>>>(RES16, ln2w, ln2b, nullptr, nullptr, out, out_scale);
}
